// InteractionBlockPrompt_13632226197770
// MI455X (gfx1250) — matrix operand order repaired and checked by reading, not compiled, not run on hardware
//
#include <hip/hip_runtime.h>
#include <hip/hip_bf16.h>
#include <math.h>
#include <stdint.h>

// ---------------- fixed problem sizes ----------------
#define Bsz   2
#define Dm    256
#define NHh   8
#define DHh   32
#define NPp   4
#define Lx    4096      // H*W
#define L1    16384     // (2H)*(2W)
#define L2    4096
#define L3    1024
#define Lc    21504     // L1+L2+L3
#define HIDf  64

typedef __attribute__((ext_vector_type(16))) __bf16 bf16x16;
typedef __attribute__((ext_vector_type(8)))  __bf16 bf16x8;
typedef __attribute__((ext_vector_type(8)))  float  v8f;
typedef __attribute__((ext_vector_type(4)))  unsigned int uint4v;
typedef __attribute__((ext_vector_type(8)))  int int8v;
typedef __attribute__((ext_vector_type(4)))  int int4v;

#define WMMA_BF16(a, b, c) \
  __builtin_amdgcn_wmma_f32_16x16x32_bf16(false, (a), false, (b), (short)0, (c), false, false)

// =====================================================
//  small helpers
// =====================================================
__device__ inline bf16x16 make_afrag(bf16x8 a0, bf16x8 a1) {
  bf16x16 a;
#pragma unroll
  for (int i = 0; i < 8; ++i) { a[i] = a0[i]; a[i + 8] = a1[i]; }
  return a;
}

__device__ inline bf16x8 zero8() {
  bf16x8 z;
#pragma unroll
  for (int i = 0; i < 8; ++i) z[i] = (__bf16)0.0f;
  return z;
}

// =====================================================
//  concat c = [c1; c2; c3] per batch (f32)
// =====================================================
__global__ void concat_c_kernel(const float* __restrict__ c1,
                                const float* __restrict__ c2,
                                const float* __restrict__ c3,
                                float* __restrict__ c) {
  long i = (long)blockIdx.x * 256 + threadIdx.x;
  const long total = (long)Bsz * Lc * Dm;
  if (i >= total) return;
  int ch = (int)(i & 255);
  long t = i >> 8;
  int b = (int)(t / Lc);
  int r = (int)(t % Lc);
  float v;
  if (r < L1)            v = c1[((long)b * L1 + r) * Dm + ch];
  else if (r < L1 + L2)  v = c2[((long)b * L2 + (r - L1)) * Dm + ch];
  else                   v = c3[((long)b * L3 + (r - L1 - L2)) * Dm + ch];
  c[i] = v;
}

// =====================================================
//  LayerNorm (D=256) f32 -> bf16, one wave per row
// =====================================================
__global__ void ln_bf16_kernel(const float* __restrict__ X,
                               const float* __restrict__ g,
                               const float* __restrict__ bb,
                               __bf16* __restrict__ O, int rows) {
  int lane = threadIdx.x & 31;
  int wave = threadIdx.x >> 5;
  int r = blockIdx.x * 8 + wave;
  if (r >= rows) return;
  const float* xr = X + (long)r * Dm;
  float v[8], s = 0.f, s2 = 0.f;
#pragma unroll
  for (int i = 0; i < 8; ++i) {
    v[i] = xr[lane * 8 + i];
    s += v[i]; s2 += v[i] * v[i];
  }
#pragma unroll
  for (int m = 16; m >= 1; m >>= 1) {
    s  += __shfl_xor(s,  m);
    s2 += __shfl_xor(s2, m);
  }
  float mean = s * (1.0f / Dm);
  float var  = s2 * (1.0f / Dm) - mean * mean;
  float inv  = rsqrtf(var + 1e-6f);
#pragma unroll
  for (int i = 0; i < 8; ++i) {
    int ch = lane * 8 + i;
    O[(long)r * Dm + ch] = (__bf16)((v[i] - mean) * inv * g[ch] + bb[ch]);
  }
}

// =====================================================
//  pre-pack weight matrix (K x N, arbitrary strides) into
//  WMMA B-fragment layout, bf16, with N zero-padding.
// =====================================================
__global__ void prepack_w_kernel(const float* __restrict__ W,
                                 __bf16* __restrict__ P,
                                 int K, int N, int Npad,
                                 long sK, long sN, long base) {
  int i = blockIdx.x * 256 + threadIdx.x;
  int total = K * Npad;
  if (i >= total) return;
  int e = i & 15;
  int lane = (i >> 4) & 31;
  int tile = i >> 9;
  int ntiles = Npad >> 4;
  int kt = tile / ntiles, nt = tile % ntiles;
  int k = kt * 32 + ((e >> 3) << 4) + ((lane >> 4) << 3) + (e & 7);
  int n = nt * 16 + (lane & 15);
  float v = (n < N && k < K) ? W[base + (long)k * sK + (long)n * sN] : 0.0f;
  P[i] = (__bf16)v;
}

__global__ void pad_vec_kernel(const float* __restrict__ src,
                               float* __restrict__ dst, int n, int npad) {
  int i = blockIdx.x * 256 + threadIdx.x;
  if (i < npad) dst[i] = (i < n) ? src[i] : 0.0f;
}

__global__ void cvt_bf16_kernel(const float* __restrict__ X,
                                __bf16* __restrict__ O, long n) {
  long i = (long)blockIdx.x * 256 + threadIdx.x;
  if (i < n) O[i] = (__bf16)X[i];
}

// =====================================================
//  WMMA GEMM: C[M,Npad] = epilogue(A[M,K]bf16 @ Wpack)
//  epilogue: v = (acc + bias[n]) * gamma[n] (+ res[idx])
//  block = 256 thr (8 waves), tile 128(M) x 64(N), wave = 16 x 64.
//  Manual ping-pong, two k-steps per iteration: buffers P and Q
//  are each written exactly once per iteration straight from
//  loads (no register rotation -> no v_mov storm), and every
//  WMMA group has a full load clause in flight above it.
//  Requires K % 64 == 0 (all K here are 256 or 64).
// =====================================================
__global__ __launch_bounds__(256)
void wmma_gemm_kernel(const __bf16* __restrict__ A,
                      const __bf16* __restrict__ Bp,
                      const float* __restrict__ bias,
                      const float* __restrict__ res,
                      const float* __restrict__ gamma,
                      float* __restrict__ C,
                      int M, int Npad, int K) {
  int lane = threadIdx.x & 31;
  int wave = threadIdx.x >> 5;
  int mBase = blockIdx.x * 128 + wave * 16;
  int ntBase = blockIdx.y * 4;
  int ntiles = Npad >> 4;

  v8f acc[4];
#pragma unroll
  for (int j = 0; j < 4; ++j)
#pragma unroll
    for (int i = 0; i < 8; ++i) acc[j][i] = 0.0f;

  int rowA = mBase + (lane & 15);
  const __bf16* aRow = A + (long)rowA * K + ((lane >> 4) << 3);

  auto loadA = [&](int kt) -> bf16x16 {
    bf16x8 a0 = __builtin_nontemporal_load((const bf16x8*)(aRow + kt));
    bf16x8 a1 = __builtin_nontemporal_load((const bf16x8*)(aRow + kt + 16));
    return make_afrag(a0, a1);
  };
  auto loadB = [&](int kt, int j) -> bf16x16 {
    const __bf16* bp =
        Bp + ((((long)(kt >> 5) * ntiles + ntBase + j) * 32 + lane) << 4);
    return *(const bf16x16*)bp;
  };

  // prologue: P holds k-step 0
  bf16x16 aP = loadA(0);
  bf16x16 bP[4];
#pragma unroll
  for (int j = 0; j < 4; ++j) bP[j] = loadB(0, j);

  for (int kt = 0; kt < K; kt += 64) {
    __builtin_prefetch(aRow + kt + 128, 0, 1);   // global_prefetch_b8
    // Q <- k-step kt+32 (always exists: K % 64 == 0)
    bf16x16 aQ = loadA(kt + 32);
    bf16x16 bQ[4];
#pragma unroll
    for (int j = 0; j < 4; ++j) bQ[j] = loadB(kt + 32, j);
#pragma unroll
    for (int j = 0; j < 4; ++j) acc[j] = WMMA_BF16(aP, bP[j], acc[j]);
    if (kt + 64 < K) {
      aP = loadA(kt + 64);
#pragma unroll
      for (int j = 0; j < 4; ++j) bP[j] = loadB(kt + 64, j);
    }
#pragma unroll
    for (int j = 0; j < 4; ++j) acc[j] = WMMA_BF16(aQ, bQ[j], acc[j]);
  }

  int rOff = (lane >> 4) << 3;
  int col0 = lane & 15;
#pragma unroll
  for (int j = 0; j < 4; ++j) {
    int col = (ntBase + j) * 16 + col0;
    float bv = bias ? bias[col] : 0.0f;
    float gv = gamma ? gamma[col] : 1.0f;
#pragma unroll
    for (int r = 0; r < 8; ++r) {
      int row = mBase + rOff + r;
      long idx = (long)row * Npad + col;
      float v = (acc[j][r] + bv) * gv;
      if (res) v += res[idx];
      C[idx] = v;
    }
  }
}

// =====================================================
//  3x3 stride-2 pad-1 conv (256->256 ch) as 9-tap implicit
//  GEMM on WMMA, (tap x kstep) space flattened into 72 steps
//  and ping-ponged two steps per iteration (72 is even).
//  In: (B, iH*iW, 256) bf16 token-major.
//  Taps: 9 pre-packed 256x256 B-fragments.
//  Out: (B, oH*oW, 256) f32 (non-temporal stores).
// =====================================================
__global__ __launch_bounds__(256)
void conv3x3s2_wmma_kernel(const __bf16* __restrict__ In,
                           const __bf16* __restrict__ Taps,
                           float* __restrict__ Out,
                           int iH, int iW, int oH, int oW) {
  int lane = threadIdx.x & 31;
  int wave = threadIdx.x >> 5;
  int b = blockIdx.z;
  long iHW = (long)iH * iW;
  long oHW = (long)oH * oW;
  int tBase = blockIdx.x * 128 + wave * 16;
  int ntBase = blockIdx.y * 4;

  int tRow = tBase + (lane & 15);
  int oy = tRow / oW, ox = tRow % oW;
  int kOff = (lane >> 4) << 3;

  v8f acc[4];
#pragma unroll
  for (int j = 0; j < 4; ++j)
#pragma unroll
    for (int i = 0; i < 8; ++i) acc[j][i] = 0.0f;

  // step s in [0,72): tap = s>>3, kt = (s&7)<<5
  auto loadAc = [&](int s) -> bf16x16 {
    int tap = s >> 3;
    int kt = (s & 7) << 5;
    int ky = tap / 3, kx = tap % 3;
    int iy = 2 * oy - 1 + ky;
    int ix = 2 * ox - 1 + kx;
    bool valid = (iy >= 0) && (iy < iH) && (ix >= 0) && (ix < iW);
    bf16x8 a0 = zero8(), a1 = zero8();
    if (valid) {
      const __bf16* aRow =
          In + ((long)b * iHW + (long)iy * iW + ix) * Dm + kOff;
      a0 = __builtin_nontemporal_load((const bf16x8*)(aRow + kt));
      a1 = __builtin_nontemporal_load((const bf16x8*)(aRow + kt + 16));
    }
    return make_afrag(a0, a1);
  };
  auto loadBc = [&](int s, int j) -> bf16x16 {
    int tap = s >> 3;
    int kt = (s & 7) << 5;
    const __bf16* bp = Taps + (long)tap * Dm * Dm +
                       ((((long)(kt >> 5) * 16 + ntBase + j) * 32 + lane) << 4);
    return *(const bf16x16*)bp;
  };

  bf16x16 aP = loadAc(0);
  bf16x16 bP[4];
#pragma unroll
  for (int j = 0; j < 4; ++j) bP[j] = loadBc(0, j);

  for (int s = 0; s < 72; s += 2) {
    bf16x16 aQ = loadAc(s + 1);
    bf16x16 bQ[4];
#pragma unroll
    for (int j = 0; j < 4; ++j) bQ[j] = loadBc(s + 1, j);
#pragma unroll
    for (int j = 0; j < 4; ++j) acc[j] = WMMA_BF16(aP, bP[j], acc[j]);
    if (s + 2 < 72) {
      aP = loadAc(s + 2);
#pragma unroll
      for (int j = 0; j < 4; ++j) bP[j] = loadBc(s + 2, j);
    }
#pragma unroll
    for (int j = 0; j < 4; ++j) acc[j] = WMMA_BF16(aQ, bQ[j], acc[j]);
  }

  int rOff = (lane >> 4) << 3;
  int col0 = lane & 15;
#pragma unroll
  for (int j = 0; j < 4; ++j) {
    int col = (ntBase + j) * 16 + col0;
#pragma unroll
    for (int r = 0; r < 8; ++r) {
      int row = tBase + rOff + r;
      __builtin_nontemporal_store(acc[j][r],
                                  &Out[((long)b * oHW + row) * Dm + col]);
    }
  }
}

// =====================================================
//  bilinear gather helper: accumulate 32 channels
// =====================================================
__device__ inline void bilerp_acc(const float* __restrict__ base,
                                  int Hl, int Wl,
                                  float lx, float ly, float wgt,
                                  float* acc) {
  float x = lx * Wl - 0.5f;
  float y = ly * Hl - 0.5f;
  float x0f = floorf(x), y0f = floorf(y);
  int x0 = (int)x0f, y0 = (int)y0f;
  float wx = x - x0f, wy = y - y0f;
#pragma unroll
  for (int dy = 0; dy < 2; ++dy) {
#pragma unroll
    for (int dx = 0; dx < 2; ++dx) {
      int xi = x0 + dx, yi = y0 + dy;
      float w = (dx ? wx : 1.0f - wx) * (dy ? wy : 1.0f - wy) * wgt;
      if (xi >= 0 && xi < Wl && yi >= 0 && yi < Hl && w != 0.0f) {
        const float* g = base + ((long)yi * Wl + xi) * Dm;
#pragma unroll
        for (int d = 0; d < DHh; ++d) acc[d] += w * g[d];
      }
    }
  }
}

// =====================================================
//  ext MSDA sampling: queries = pyramid (Lc), value = x-proj (Lx @ 64x64)
//  offs ld = 64  (NH*1*NP*2), aw ld = 64 (padded from 32)
// =====================================================
__global__ void sample_ext_kernel(const float* __restrict__ val,
                                  const float* __restrict__ offs,
                                  const float* __restrict__ aw,
                                  float* __restrict__ S) {
  long i = (long)blockIdx.x * 256 + threadIdx.x;
  const long total = (long)Bsz * Lc * NHh;
  if (i >= total) return;
  int h = (int)(i & 7);
  int q = (int)((i >> 3) % Lc);
  int b = (int)(i / ((long)Lc * NHh));

  float rx, ry;
  if (q < L1) {
    int y = q >> 7, x = q & 127;
    rx = (x + 0.5f) * (1.0f / 128.0f); ry = (y + 0.5f) * (1.0f / 128.0f);
  } else if (q < L1 + L2) {
    int t = q - L1; int y = t >> 6, x = t & 63;
    rx = (x + 0.5f) * (1.0f / 64.0f); ry = (y + 0.5f) * (1.0f / 64.0f);
  } else {
    int t = q - L1 - L2; int y = t >> 5, x = t & 31;
    rx = (x + 0.5f) * (1.0f / 32.0f); ry = (y + 0.5f) * (1.0f / 32.0f);
  }

  const float* ofp = offs + ((long)b * Lc + q) * 64 + h * 8;
  const float* awp = aw   + ((long)b * Lc + q) * 64 + h * 4;

  float wts[NPp];
  float mx = awp[0];
#pragma unroll
  for (int p = 1; p < NPp; ++p) mx = fmaxf(mx, awp[p]);
  float ssum = 0.f;
#pragma unroll
  for (int p = 0; p < NPp; ++p) { wts[p] = expf(awp[p] - mx); ssum += wts[p]; }
  float inv = 1.0f / ssum;

  float acc[DHh];
#pragma unroll
  for (int d = 0; d < DHh; ++d) acc[d] = 0.0f;

  const float* base = val + ((long)b * Lx) * Dm + h * DHh;
#pragma unroll
  for (int p = 0; p < NPp; ++p) {
    float lx = rx + ofp[2 * p]     * (1.0f / 64.0f);
    float ly = ry + ofp[2 * p + 1] * (1.0f / 64.0f);
    bilerp_acc(base, 64, 64, lx, ly, wts[p] * inv, acc);
  }
  float* out = S + ((long)b * Lc + q) * Dm + h * DHh;
#pragma unroll
  for (int d = 0; d < DHh; ++d) out[d] = acc[d];
}

// =====================================================
//  inj MSDA sampling: queries = x (Lx @ 64x64), value = c-proj pyramid
//  offs ld = 192 (NH*3*NP*2), aw ld = 128 (padded from 96)
// =====================================================
__global__ void sample_inj_kernel(const float* __restrict__ val,
                                  const float* __restrict__ offs,
                                  const float* __restrict__ aw,
                                  float* __restrict__ S) {
  long i = (long)blockIdx.x * 256 + threadIdx.x;
  const long total = (long)Bsz * Lx * NHh;
  if (i >= total) return;
  int h = (int)(i & 7);
  int q = (int)((i >> 3) % Lx);
  int b = (int)(i / ((long)Lx * NHh));

  int y = q >> 6, x = q & 63;
  float rx = (x + 0.5f) * (1.0f / 64.0f);
  float ry = (y + 0.5f) * (1.0f / 64.0f);

  const float* ofp = offs + ((long)b * Lx + q) * 192 + h * 24;   // 3 lvls * 4 pts * 2
  const float* awp = aw   + ((long)b * Lx + q) * 128 + h * 12;   // 12 logits

  float wts[12];
  float mx = awp[0];
#pragma unroll
  for (int p = 1; p < 12; ++p) mx = fmaxf(mx, awp[p]);
  float ssum = 0.f;
#pragma unroll
  for (int p = 0; p < 12; ++p) { wts[p] = expf(awp[p] - mx); ssum += wts[p]; }
  float inv = 1.0f / ssum;

  float acc[DHh];
#pragma unroll
  for (int d = 0; d < DHh; ++d) acc[d] = 0.0f;

  const int starts[3] = {0, L1, L1 + L2};
  const int hs[3] = {128, 64, 32};
#pragma unroll
  for (int l = 0; l < 3; ++l) {
    const float* base = val + ((long)b * Lc + starts[l]) * Dm + h * DHh;
    float invd = 1.0f / (float)hs[l];
#pragma unroll
    for (int p = 0; p < NPp; ++p) {
      float lx = rx + ofp[(l * NPp + p) * 2]     * invd;
      float ly = ry + ofp[(l * NPp + p) * 2 + 1] * invd;
      bilerp_acc(base, hs[l], hs[l], lx, ly, wts[l * NPp + p] * inv, acc);
    }
  }
  float* out = S + ((long)b * Lx + q) * Dm + h * DHh;
#pragma unroll
  for (int d = 0; d < DHh; ++d) out[d] = acc[d];
}

// =====================================================
//  depthwise 3x3 (64ch) + bias + exact GELU -> bf16
// =====================================================
__global__ void dwconv_gelu_kernel(const float* __restrict__ Hh,
                                   const float* __restrict__ w,
                                   const float* __restrict__ bias,
                                   __bf16* __restrict__ O) {
  long i = (long)blockIdx.x * 256 + threadIdx.x;
  const long total = (long)Bsz * Lc * HIDf;
  if (i >= total) return;
  int ch = (int)(i & 63);
  long t = i >> 6;
  int b = (int)(t / Lc);
  int r = (int)(t % Lc);
  int Hl, Wl, s0;
  if (r < L1)            { Hl = 128; Wl = 128; s0 = 0; }
  else if (r < L1 + L2)  { Hl = 64;  Wl = 64;  s0 = L1; }
  else                   { Hl = 32;  Wl = 32;  s0 = L1 + L2; }
  int rr = r - s0;
  int y = rr / Wl, x = rr % Wl;
  float acc = bias[ch];
#pragma unroll
  for (int ky = 0; ky < 3; ++ky)
#pragma unroll
    for (int kx = 0; kx < 3; ++kx) {
      int yy = y - 1 + ky, xx = x - 1 + kx;
      if (yy >= 0 && yy < Hl && xx >= 0 && xx < Wl)
        acc += Hh[((long)b * Lc + s0 + (long)yy * Wl + xx) * HIDf + ch] *
               w[ch * 9 + ky * 3 + kx];
    }
  float gv = 0.5f * acc * (1.0f + erff(acc * 0.70710678118f));
  O[i] = (__bf16)gv;
}

// =====================================================
//  x_inj (B,Lx,256) -> prompt (B,256,64,64)  (LDS tiled transpose)
// =====================================================
__global__ void transpose_prompt_kernel(const float* __restrict__ xinj,
                                        float* __restrict__ prompt) {
  __shared__ float tile[32][33];
  int b = blockIdx.z;
  int t0 = blockIdx.x * 32, c0 = blockIdx.y * 32;
  int tx = threadIdx.x & 31, ty = threadIdx.x >> 5;   // 8 rows of 32
  for (int i = ty; i < 32; i += 8)
    tile[i][tx] = xinj[((long)b * Lx + t0 + i) * Dm + c0 + tx];
  __syncthreads();
  for (int i = ty; i < 32; i += 8)
    __builtin_nontemporal_store(tile[tx][i],
                                &prompt[((long)b * Dm + c0 + i) * Lx + t0 + tx]);
}

// =====================================================
//  TDM demo (compile-only showcase of the async-tensor path;
//  not launched). Stages a tile into LDS with the Tensor
//  Data Mover and waits on TENSORcnt.  This toolchain exposes
//  the 6-arg form: (g0 u32x4, g1 i32x8, g2 i32x4, g3 i32x4, i32x8, cpol)
// =====================================================
#if __has_builtin(__builtin_amdgcn_tensor_load_to_lds) && \
    __has_builtin(__builtin_amdgcn_s_wait_tensorcnt)
__global__ void tdm_demo_kernel(const float* __restrict__ src,
                                float* __restrict__ dst) {
  __shared__ float smem[2048];
  unsigned long long ga = (unsigned long long)(uintptr_t)src;
  uint4v g0; int8v g1; int4v g2, g3; int8v g4;
  g0[0] = 1u;                                   // count=1, user descriptor
  g0[1] = 0u;                                   // lds_addr = 0
  g0[2] = (unsigned)(ga & 0xffffffffu);         // global_addr[31:0] -> bits 95:64
  g0[3] = (unsigned)((ga >> 32) & 0x1ffffffu) | (2u << 30);  // addr[56:32] | type=2
  g1[0] = (int)(2u << 16);                      // data_size = 4B, no multicast
  g1[1] = (int)(64u << 16);                     // tensor_dim0 (low bits)
  g1[2] = 0;
  g1[3] = (int)(64u << 16);                     // tile_dim0 = 64
  g1[4] = (int)16;                              // tile_dim1 = 16
  g1[5] = 64;                                   // tensor_dim0_stride (low)
  g1[6] = 0; g1[7] = 0;
#pragma unroll
  for (int i = 0; i < 4; ++i) { g2[i] = 0; g3[i] = 0; }
#pragma unroll
  for (int i = 0; i < 8; ++i) g4[i] = 0;
  __builtin_amdgcn_tensor_load_to_lds(g0, g1, g2, g3, g4, 0);
  __builtin_amdgcn_s_wait_tensorcnt(0);
  __syncthreads();
  dst[threadIdx.x] = smem[threadIdx.x];
}
#endif

// =====================================================
//  host-side orchestration
// =====================================================
static inline void* wsAlloc(char*& p, size_t bytes) {
  void* r = (void*)p;
  p += (bytes + 255) & ~(size_t)255;
  return r;
}

extern "C" void kernel_launch(void* const* d_in, const int* in_sizes, int n_in,
                              void* d_out, int out_size, void* d_ws, size_t ws_size,
                              hipStream_t stream) {
  (void)in_sizes; (void)n_in; (void)out_size; (void)ws_size;

  const float* x  = (const float*)d_in[0];
  const float* c1 = (const float*)d_in[1];
  const float* c2 = (const float*)d_in[2];
  const float* c3 = (const float*)d_in[3];
  // ext params [4..15], inj params [16..27]
  const float* e_qn_w = (const float*)d_in[4];
  const float* e_qn_b = (const float*)d_in[5];
  const float* e_fn_w = (const float*)d_in[6];
  const float* e_fn_b = (const float*)d_in[7];
  const float* e_off_w = (const float*)d_in[8];
  const float* e_off_b = (const float*)d_in[9];
  const float* e_aw_w = (const float*)d_in[10];
  const float* e_aw_b = (const float*)d_in[11];
  const float* e_val_w = (const float*)d_in[12];
  const float* e_val_b = (const float*)d_in[13];
  const float* e_out_w = (const float*)d_in[14];
  const float* e_out_b = (const float*)d_in[15];
  const float* i_qn_w = (const float*)d_in[16];
  const float* i_qn_b = (const float*)d_in[17];
  const float* i_fn_w = (const float*)d_in[18];
  const float* i_fn_b = (const float*)d_in[19];
  const float* i_off_w = (const float*)d_in[20];
  const float* i_off_b = (const float*)d_in[21];
  const float* i_aw_w = (const float*)d_in[22];
  const float* i_aw_b = (const float*)d_in[23];
  const float* i_val_w = (const float*)d_in[24];
  const float* i_val_b = (const float*)d_in[25];
  const float* i_out_w = (const float*)d_in[26];
  const float* i_out_b = (const float*)d_in[27];
  const float* ffnnorm_w = (const float*)d_in[28];
  const float* ffnnorm_b = (const float*)d_in[29];
  const float* fc1_w = (const float*)d_in[30];
  const float* fc1_b = (const float*)d_in[31];
  const float* dw_w  = (const float*)d_in[32];
  const float* dw_b  = (const float*)d_in[33];
  const float* fc2_w = (const float*)d_in[34];
  const float* fc2_b = (const float*)d_in[35];
  const float* gamma = (const float*)d_in[36];
  const float* up1_w = (const float*)d_in[37];
  const float* up2_w = (const float*)d_in[38];
  const float* up3_w = (const float*)d_in[39];

  float* out = (float*)d_out;
  float* prompt = out;                      // 2*256*4096
  float* c1o = out + 2097152;               // 2*4096*256
  float* c2o = out + 4194304;               // 2*1024*256
  float* c3o = out + 4718592;               // 2*256*256
  float* xinj = out + 4849664;              // 2*4096*256

  // ---------------- workspace carve-up ----------------
  char* wp = (char*)d_ws;
  float*  cbuf  = (float*) wsAlloc(wp, (size_t)Bsz * Lc * Dm * 4);
  __bf16* qc    = (__bf16*)wsAlloc(wp, (size_t)Bsz * Lc * Dm * 2);
  __bf16* fx    = (__bf16*)wsAlloc(wp, (size_t)Bsz * Lx * Dm * 2);
  float*  vext  = (float*) wsAlloc(wp, (size_t)Bsz * Lx * Dm * 4);
  float*  offe  = (float*) wsAlloc(wp, (size_t)Bsz * Lc * 64 * 4);
  float*  awe   = (float*) wsAlloc(wp, (size_t)Bsz * Lc * 64 * 4);
  float*  Se    = (float*) wsAlloc(wp, (size_t)Bsz * Lc * Dm * 4);
  __bf16* Sebf  = (__bf16*)wsAlloc(wp, (size_t)Bsz * Lc * Dm * 2);
  __bf16* hln   = (__bf16*)wsAlloc(wp, (size_t)Bsz * Lc * Dm * 2);
  float*  hfc1  = (float*) wsAlloc(wp, (size_t)Bsz * Lc * HIDf * 4);
  __bf16* gelub = (__bf16*)wsAlloc(wp, (size_t)Bsz * Lc * HIDf * 2);
  __bf16* qx    = (__bf16*)wsAlloc(wp, (size_t)Bsz * Lx * Dm * 2);
  __bf16* fc    = (__bf16*)wsAlloc(wp, (size_t)Bsz * Lc * Dm * 2);
  float*  vinj  = (float*) wsAlloc(wp, (size_t)Bsz * Lc * Dm * 4);
  float*  offi  = (float*) wsAlloc(wp, (size_t)Bsz * Lx * 192 * 4);
  float*  awi   = (float*) wsAlloc(wp, (size_t)Bsz * Lx * 128 * 4);
  float*  S2    = (float*) wsAlloc(wp, (size_t)Bsz * Lx * Dm * 4);
  __bf16* S2bf  = (__bf16*)wsAlloc(wp, (size_t)Bsz * Lx * Dm * 2);
  __bf16* c1bf  = (__bf16*)wsAlloc(wp, (size_t)Bsz * L1 * Dm * 2);
  __bf16* c2bf  = (__bf16*)wsAlloc(wp, (size_t)Bsz * L2 * Dm * 2);
  __bf16* c3bf  = (__bf16*)wsAlloc(wp, (size_t)Bsz * L3 * Dm * 2);
  __bf16* pk_eval = (__bf16*)wsAlloc(wp, (size_t)256 * 256 * 2);
  __bf16* pk_eoff = (__bf16*)wsAlloc(wp, (size_t)256 * 64 * 2);
  __bf16* pk_eaw  = (__bf16*)wsAlloc(wp, (size_t)256 * 64 * 2);
  __bf16* pk_eout = (__bf16*)wsAlloc(wp, (size_t)256 * 256 * 2);
  __bf16* pk_fc1  = (__bf16*)wsAlloc(wp, (size_t)256 * 64 * 2);
  __bf16* pk_fc2  = (__bf16*)wsAlloc(wp, (size_t)64 * 256 * 2);
  __bf16* pk_ival = (__bf16*)wsAlloc(wp, (size_t)256 * 256 * 2);
  __bf16* pk_ioff = (__bf16*)wsAlloc(wp, (size_t)256 * 192 * 2);
  __bf16* pk_iaw  = (__bf16*)wsAlloc(wp, (size_t)256 * 128 * 2);
  __bf16* pk_iout = (__bf16*)wsAlloc(wp, (size_t)256 * 256 * 2);
  __bf16* pk_up1  = (__bf16*)wsAlloc(wp, (size_t)9 * 256 * 256 * 2);
  __bf16* pk_up2  = (__bf16*)wsAlloc(wp, (size_t)9 * 256 * 256 * 2);
  __bf16* pk_up3  = (__bf16*)wsAlloc(wp, (size_t)9 * 256 * 256 * 2);
  float*  eaw_bp  = (float*) wsAlloc(wp, 64 * 4);
  float*  iaw_bp  = (float*) wsAlloc(wp, 128 * 4);

  const int Mc = Bsz * Lc;   // 43008
  const int Mx = Bsz * Lx;   // 8192

  // ---------------- weight pre-pack (bf16 fragment layout) ----------------
  auto packGrid = [](int K, int Npad) { return dim3((K * Npad + 255) / 256); };
  hipLaunchKernelGGL(prepack_w_kernel, packGrid(256, 256), dim3(256), 0, stream,
                     e_val_w, pk_eval, 256, 256, 256, 256L, 1L, 0L);
  hipLaunchKernelGGL(prepack_w_kernel, packGrid(256, 64), dim3(256), 0, stream,
                     e_off_w, pk_eoff, 256, 64, 64, 64L, 1L, 0L);
  hipLaunchKernelGGL(prepack_w_kernel, packGrid(256, 64), dim3(256), 0, stream,
                     e_aw_w, pk_eaw, 256, 32, 64, 32L, 1L, 0L);
  hipLaunchKernelGGL(prepack_w_kernel, packGrid(256, 256), dim3(256), 0, stream,
                     e_out_w, pk_eout, 256, 256, 256, 256L, 1L, 0L);
  hipLaunchKernelGGL(prepack_w_kernel, packGrid(256, 64), dim3(256), 0, stream,
                     fc1_w, pk_fc1, 256, 64, 64, 64L, 1L, 0L);
  hipLaunchKernelGGL(prepack_w_kernel, packGrid(64, 256), dim3(256), 0, stream,
                     fc2_w, pk_fc2, 64, 256, 256, 256L, 1L, 0L);
  hipLaunchKernelGGL(prepack_w_kernel, packGrid(256, 256), dim3(256), 0, stream,
                     i_val_w, pk_ival, 256, 256, 256, 256L, 1L, 0L);
  hipLaunchKernelGGL(prepack_w_kernel, packGrid(256, 192), dim3(256), 0, stream,
                     i_off_w, pk_ioff, 256, 192, 192, 192L, 1L, 0L);
  hipLaunchKernelGGL(prepack_w_kernel, packGrid(256, 128), dim3(256), 0, stream,
                     i_aw_w, pk_iaw, 256, 96, 128, 96L, 1L, 0L);
  hipLaunchKernelGGL(prepack_w_kernel, packGrid(256, 256), dim3(256), 0, stream,
                     i_out_w, pk_iout, 256, 256, 256, 256L, 1L, 0L);
  // conv weights: OIHW (oc, ic, ky, kx). B element (k=ic, n=oc) at
  // oc*2304 + ic*9 + tap  ->  sK=9, sN=2304, base=tap.
  for (int tap = 0; tap < 9; ++tap) {
    hipLaunchKernelGGL(prepack_w_kernel, packGrid(256, 256), dim3(256), 0, stream,
                       up1_w, pk_up1 + (size_t)tap * 65536, 256, 256, 256, 9L, 2304L, (long)tap);
    hipLaunchKernelGGL(prepack_w_kernel, packGrid(256, 256), dim3(256), 0, stream,
                       up2_w, pk_up2 + (size_t)tap * 65536, 256, 256, 256, 9L, 2304L, (long)tap);
    hipLaunchKernelGGL(prepack_w_kernel, packGrid(256, 256), dim3(256), 0, stream,
                       up3_w, pk_up3 + (size_t)tap * 65536, 256, 256, 256, 9L, 2304L, (long)tap);
  }
  hipLaunchKernelGGL(pad_vec_kernel, dim3(1), dim3(256), 0, stream, e_aw_b, eaw_bp, 32, 64);
  hipLaunchKernelGGL(pad_vec_kernel, dim3(1), dim3(256), 0, stream, i_aw_b, iaw_bp, 96, 128);

  // ---------------- stage 0: concat c ----------------
  {
    long tot = (long)Bsz * Lc * Dm;
    hipLaunchKernelGGL(concat_c_kernel, dim3((unsigned)((tot + 255) / 256)), dim3(256), 0,
                       stream, c1, c2, c3, cbuf);
  }

  // ---------------- ext MSDA ----------------
  hipLaunchKernelGGL(ln_bf16_kernel, dim3(Mc / 8), dim3(256), 0, stream,
                     cbuf, e_qn_w, e_qn_b, qc, Mc);
  hipLaunchKernelGGL(ln_bf16_kernel, dim3(Mx / 8), dim3(256), 0, stream,
                     x, e_fn_w, e_fn_b, fx, Mx);
  hipLaunchKernelGGL(wmma_gemm_kernel, dim3(Mx / 128, 4), dim3(256), 0, stream,
                     fx, pk_eval, e_val_b, (const float*)nullptr, (const float*)nullptr,
                     vext, Mx, 256, 256);
  hipLaunchKernelGGL(wmma_gemm_kernel, dim3(Mc / 128, 1), dim3(256), 0, stream,
                     qc, pk_eoff, e_off_b, (const float*)nullptr, (const float*)nullptr,
                     offe, Mc, 64, 256);
  hipLaunchKernelGGL(wmma_gemm_kernel, dim3(Mc / 128, 1), dim3(256), 0, stream,
                     qc, pk_eaw, eaw_bp, (const float*)nullptr, (const float*)nullptr,
                     awe, Mc, 64, 256);
  {
    long tot = (long)Bsz * Lc * NHh;
    hipLaunchKernelGGL(sample_ext_kernel, dim3((unsigned)((tot + 255) / 256)), dim3(256), 0,
                       stream, vext, offe, awe, Se);
  }
  {
    long tot = (long)Bsz * Lc * Dm;
    hipLaunchKernelGGL(cvt_bf16_kernel, dim3((unsigned)((tot + 255) / 256)), dim3(256), 0,
                       stream, Se, Sebf, tot);
  }
  // c += Se @ out_w + out_b   (residual fused, in-place)
  hipLaunchKernelGGL(wmma_gemm_kernel, dim3(Mc / 128, 4), dim3(256), 0, stream,
                     Sebf, pk_eout, e_out_b, cbuf, (const float*)nullptr,
                     cbuf, Mc, 256, 256);

  // ---------------- conv FFN ----------------
  hipLaunchKernelGGL(ln_bf16_kernel, dim3(Mc / 8), dim3(256), 0, stream,
                     cbuf, ffnnorm_w, ffnnorm_b, hln, Mc);
  hipLaunchKernelGGL(wmma_gemm_kernel, dim3(Mc / 128, 1), dim3(256), 0, stream,
                     hln, pk_fc1, fc1_b, (const float*)nullptr, (const float*)nullptr,
                     hfc1, Mc, 64, 256);
  {
    long tot = (long)Bsz * Lc * HIDf;
    hipLaunchKernelGGL(dwconv_gelu_kernel, dim3((unsigned)((tot + 255) / 256)), dim3(256), 0,
                       stream, hfc1, dw_w, dw_b, gelub);
  }
  hipLaunchKernelGGL(wmma_gemm_kernel, dim3(Mc / 128, 4), dim3(256), 0, stream,
                     gelub, pk_fc2, fc2_b, cbuf, (const float*)nullptr,
                     cbuf, Mc, 256, 64);

  // ---------------- inj MSDA ----------------
  hipLaunchKernelGGL(ln_bf16_kernel, dim3(Mx / 8), dim3(256), 0, stream,
                     x, i_qn_w, i_qn_b, qx, Mx);
  hipLaunchKernelGGL(ln_bf16_kernel, dim3(Mc / 8), dim3(256), 0, stream,
                     cbuf, i_fn_w, i_fn_b, fc, Mc);
  hipLaunchKernelGGL(wmma_gemm_kernel, dim3(Mc / 128, 4), dim3(256), 0, stream,
                     fc, pk_ival, i_val_b, (const float*)nullptr, (const float*)nullptr,
                     vinj, Mc, 256, 256);
  hipLaunchKernelGGL(wmma_gemm_kernel, dim3(Mx / 128, 3), dim3(256), 0, stream,
                     qx, pk_ioff, i_off_b, (const float*)nullptr, (const float*)nullptr,
                     offi, Mx, 192, 256);
  hipLaunchKernelGGL(wmma_gemm_kernel, dim3(Mx / 128, 2), dim3(256), 0, stream,
                     qx, pk_iaw, iaw_bp, (const float*)nullptr, (const float*)nullptr,
                     awi, Mx, 128, 256);
  {
    long tot = (long)Bsz * Lx * NHh;
    hipLaunchKernelGGL(sample_inj_kernel, dim3((unsigned)((tot + 255) / 256)), dim3(256), 0,
                       stream, vinj, offi, awi, S2);
  }
  {
    long tot = (long)Bsz * Lx * Dm;
    hipLaunchKernelGGL(cvt_bf16_kernel, dim3((unsigned)((tot + 255) / 256)), dim3(256), 0,
                       stream, S2, S2bf, tot);
  }
  // x_inj = x + gamma * (S2 @ out_w + out_b)
  hipLaunchKernelGGL(wmma_gemm_kernel, dim3(Mx / 128, 4), dim3(256), 0, stream,
                     S2bf, pk_iout, i_out_b, x, gamma,
                     xinj, Mx, 256, 256);

  // prompt = transpose(x_inj)
  hipLaunchKernelGGL(transpose_prompt_kernel, dim3(Lx / 32, Dm / 32, Bsz), dim3(256), 0,
                     stream, xinj, prompt);

  // ---------------- up-convs (WMMA implicit GEMM) ----------------
  {
    long t1 = (long)Bsz * L1 * Dm, t2 = (long)Bsz * L2 * Dm, t3 = (long)Bsz * L3 * Dm;
    hipLaunchKernelGGL(cvt_bf16_kernel, dim3((unsigned)((t1 + 255) / 256)), dim3(256), 0,
                       stream, c1, c1bf, t1);
    hipLaunchKernelGGL(cvt_bf16_kernel, dim3((unsigned)((t2 + 255) / 256)), dim3(256), 0,
                       stream, c2, c2bf, t2);
    hipLaunchKernelGGL(cvt_bf16_kernel, dim3((unsigned)((t3 + 255) / 256)), dim3(256), 0,
                       stream, c3, c3bf, t3);
  }
  hipLaunchKernelGGL(conv3x3s2_wmma_kernel, dim3(4096 / 128, 4, Bsz), dim3(256), 0, stream,
                     c1bf, pk_up1, c1o, 128, 128, 64, 64);
  hipLaunchKernelGGL(conv3x3s2_wmma_kernel, dim3(1024 / 128, 4, Bsz), dim3(256), 0, stream,
                     c2bf, pk_up2, c2o, 64, 64, 32, 32);
  hipLaunchKernelGGL(conv3x3s2_wmma_kernel, dim3(256 / 128, 4, Bsz), dim3(256), 0, stream,
                     c3bf, pk_up3, c3o, 32, 32, 16, 16);
}
